// DEFA_54374285967519
// MI455X (gfx1250) — matrix operand order repaired and checked by reading, not compiled, not run on hardware
//
#include <hip/hip_runtime.h>
#include <stdint.h>

// ---------------- CDNA5 WMMA plumbing (wave32, gfx1250) ----------------
typedef __attribute__((ext_vector_type(16))) __bf16 v16bf;
typedef __attribute__((ext_vector_type(8)))  float  v8f;

union AF { v16bf v; uint4 u[2]; unsigned short s[16]; };
union CF { v8f  v; float f[8]; float4 q[2]; };

__device__ __forceinline__ unsigned short f2bf(float x) {
  unsigned int u = __float_as_uint(x);
  u += 0x7FFFu + ((u >> 16) & 1u);            // round-to-nearest-even
  return (unsigned short)(u >> 16);
}

#define WMMA_BF16(A, B, C) \
  __builtin_amdgcn_wmma_f32_16x16x32_bf16(false, (A), false, (B), (short)0, (C), false, false)

// Problem dims
#define BB   8
#define CC   512
#define NN   1024      // 32*32
#define NHD  8
#define HD   64
#define MTOT 8192      // B * N

// ---------------- small scalar kernels ----------------

__global__ void k_pool(const float* vi, const float* ir, float* pooled) {
  int bc = blockIdx.x;                          // 0..4095 == b*512+c
  const float* pv = vi + (size_t)bc * NN;
  const float* pi = ir + (size_t)bc * NN;
  float s = 0.f;
  for (int i = threadIdx.x; i < NN; i += 256) s += pv[i] + pi[i];
  __shared__ float red[256];
  red[threadIdx.x] = s; __syncthreads();
  for (int off = 128; off > 0; off >>= 1) {
    if (threadIdx.x < off) red[threadIdx.x] += red[threadIdx.x + off];
    __syncthreads();
  }
  if (threadIdx.x == 0) pooled[bc] = red[0] * (1.f / (float)NN);
}

__global__ void k_mlp1(const float* pooled, const float* w1, const float* b1, float* hid) {
  int idx = blockIdx.x * blockDim.x + threadIdx.x;   // b*2048 + j
  if (idx >= BB * 2048) return;
  int b = idx >> 11, j = idx & 2047;
  const float* p = pooled + b * CC;
  const float* w = w1 + (size_t)j * CC;
  float s = b1[j];
  for (int k = 0; k < CC; k++) s += p[k] * w[k];
  hid[idx] = fmaxf(s, 0.f);
}

__global__ void k_mlp2(const float* hid, const float* w2, const float* b2, float* kw) {
  int b = blockIdx.x, t = threadIdx.x;               // 32 threads
  __shared__ float sc[32], se[32];
  float s = -1e30f;
  if (t < 25) {
    const float* h = hid + b * 2048;
    const float* w = w2 + (size_t)t * 2048;
    float acc = b2[t];
    for (int k = 0; k < 2048; k++) acc += h[k] * w[k];
    s = acc;
  }
  sc[t] = s; __syncthreads();
  float m = -1e30f;
  for (int i = 0; i < 25; i++) m = fmaxf(m, sc[i]);
  float e = (t < 25) ? __expf(sc[t] - m) : 0.f;
  se[t] = e; __syncthreads();
  float sum = 0.f;
  for (int i = 0; i < 25; i++) sum += se[i];
  if (t < 25) kw[b * 25 + t] = e / sum;
}

// (b,c,n) fp32 -> (b,n,c) bf16 for both vi & ir
__global__ void k_pack_seq(const float* vi, const float* ir,
                           unsigned short* Xvi, unsigned short* Xir) {
  size_t idx = (size_t)blockIdx.x * blockDim.x + threadIdx.x;
  if (idx >= (size_t)BB * CC * NN) return;
  int n = (int)(idx & (NN - 1));
  size_t t = idx >> 10; int c = (int)(t & (CC - 1)); int b = (int)(t >> 9);
  size_t dst = ((size_t)(b * NN + n)) * CC + c;
  Xvi[dst] = f2bf(vi[idx]);
  Xir[dst] = f2bf(ir[idx]);
}

__global__ void k_cvt(const float* src, unsigned short* dst, int n) {
  int i = blockIdx.x * blockDim.x + threadIdx.x;
  if (i < n) dst[i] = f2bf(src[i]);
}

// enh_w [oc][ic][3][3] fp32 -> [tap][oc][ic] bf16
__global__ void k_pack_enhw(const float* w, unsigned short* pk) {
  size_t idx = (size_t)blockIdx.x * blockDim.x + threadIdx.x;
  if (idx >= (size_t)CC * CC * 9) return;
  int tap = (int)(idx % 9);
  size_t t = idx / 9; int ic = (int)(t & (CC - 1)); int oc = (int)(t >> 9);
  pk[((size_t)tap * CC + oc) * CC + ic] = f2bf(w[idx]);
}

// ---------------- WMMA GEMM: OUT = A(M x 512, bf16) @ W[n][k]^T + bias ----------------
// Per wave: 32(M) x 64(N) tile; 2 A-fragments, B-fragments reused across both M halves.
// MODE 0: bf16 row-major out[row*512+col]
// MODE 1: bf16 per-head-transposed Vt[((b*8+h)*64+d)*1024 + nseq]
// MODE 2: fp32 spatial out[(b*512+col)*1024 + nseq] (vectorized stores)
template <int MODE>
__global__ __launch_bounds__(256) void k_gemm(const unsigned short* __restrict__ A,
                                              const unsigned short* __restrict__ W,
                                              const float* __restrict__ bias,
                                              void* __restrict__ Out, int M) {
  int gw = (blockIdx.x * blockDim.x + threadIdx.x) >> 5;
  int lane = threadIdx.x & 31;
  int mt = gw >> 3, ng = gw & 7;                 // N=512 -> 8 groups of 64
  if (mt * 32 >= M) return;
  int hi = lane >> 4, ln = lane & 15;

  const unsigned short* arow0 = A + (size_t)(mt * 32 + ln) * CC;
  const unsigned short* arow1 = arow0 + (size_t)16 * CC;
  CF c[2][4];
#pragma unroll
  for (int h2 = 0; h2 < 2; h2++)
#pragma unroll
    for (int s = 0; s < 4; s++)
#pragma unroll
      for (int i = 0; i < 8; i++) c[h2][s].f[i] = 0.f;

  for (int kb = 0; kb < CC; kb += 32) {
    AF a0, a1;
    a0.u[0] = *(const uint4*)(arow0 + kb + 8 * hi);
    a0.u[1] = *(const uint4*)(arow0 + kb + 16 + 8 * hi);
    a1.u[0] = *(const uint4*)(arow1 + kb + 8 * hi);
    a1.u[1] = *(const uint4*)(arow1 + kb + 16 + 8 * hi);
#pragma unroll
    for (int s = 0; s < 4; s++) {
      int col = ng * 64 + s * 16 + ln;
      const unsigned short* wr = W + (size_t)col * CC + kb + 8 * hi;
      AF bf;
      bf.u[0] = *(const uint4*)(wr);
      bf.u[1] = *(const uint4*)(wr + 16);
      c[0][s].v = WMMA_BF16(a0.v, bf.v, c[0][s].v);
      c[1][s].v = WMMA_BF16(a1.v, bf.v, c[1][s].v);
    }
  }

#pragma unroll
  for (int h2 = 0; h2 < 2; h2++) {
    if (MODE == 0) {
      unsigned short* O = (unsigned short*)Out;
#pragma unroll
      for (int s = 0; s < 4; s++) {
        int col = ng * 64 + s * 16 + ln;
        float bi = bias[col];
#pragma unroll
        for (int r = 0; r < 8; r++) {
          int row = mt * 32 + h2 * 16 + r + 8 * hi;
          O[(size_t)row * CC + col] = f2bf(c[h2][s].f[r] + bi);
        }
      }
    } else if (MODE == 1) {
      unsigned short* O = (unsigned short*)Out;
#pragma unroll
      for (int s = 0; s < 4; s++) {
        int col = ng * 64 + s * 16 + ln;
        float bi = bias[col];
        int h = col >> 6, d = col & 63;
#pragma unroll
        for (int r = 0; r < 8; r++) {
          int row = mt * 32 + h2 * 16 + r + 8 * hi;
          int b = row >> 10, nn = row & (NN - 1);
          O[(((size_t)(b * NHD + h)) * HD + d) * NN + nn] = f2bf(c[h2][s].f[r] + bi);
        }
      }
    } else {
      float* O = (float*)Out;
      int row0 = mt * 32 + h2 * 16 + 8 * hi;
      int b = row0 >> 10, n0 = row0 & (NN - 1);
#pragma unroll
      for (int s = 0; s < 4; s++) {
        int col = ng * 64 + s * 16 + ln;
        float bi = bias[col];
        CF t;
#pragma unroll
        for (int r = 0; r < 8; r++) t.f[r] = c[h2][s].f[r] + bi;
        float4* dst = (float4*)(O + ((size_t)(b * CC + col)) * NN + n0);
        dst[0] = t.q[0]; dst[1] = t.q[1];
      }
    }
  }
}

// ---------------- Flash attention: one wave per (b,h,16-query tile) ----------------
__global__ __launch_bounds__(128) void k_attn(const unsigned short* __restrict__ Qb,
                                              const unsigned short* __restrict__ Kb,
                                              const unsigned short* __restrict__ Vt,
                                              unsigned short* __restrict__ ctxb) {
  __shared__ __align__(16) unsigned short pbuf[4][16][32];
  int w = threadIdx.x >> 5, lane = threadIdx.x & 31;
  int gw = blockIdx.x * 4 + w;                 // 0..4095
  int qt = gw & 63;
  int t = gw >> 6; int h = t & 7, b = t >> 3;
  int hi = lane >> 4, ln = lane & 15;

  const unsigned short* qrow = Qb + ((size_t)(b * NN + qt * 16 + ln)) * CC + h * HD;
  AF aq0, aq1;
  aq0.u[0] = *(const uint4*)(qrow + 16 * hi);
  aq0.u[1] = *(const uint4*)(qrow + 16 * hi + 8);
  aq1.u[0] = *(const uint4*)(qrow + 32 + 16 * hi);
  aq1.u[1] = *(const uint4*)(qrow + 32 + 16 * hi + 8);

  CF o[4];
  float rmax[8], rsum[8];
#pragma unroll
  for (int r = 0; r < 8; r++) { rmax[r] = -1e30f; rsum[r] = 0.f; }
#pragma unroll
  for (int d = 0; d < 4; d++)
#pragma unroll
    for (int i = 0; i < 8; i++) o[d].f[i] = 0.f;

  for (int kc = 0; kc < 32; kc++) {
    int k0 = kc * 32;
    CF s0, s1, z;
#pragma unroll
    for (int i = 0; i < 8; i++) z.f[i] = 0.f;
    {
      const unsigned short* kr = Kb + ((size_t)(b * NN + k0 + ln)) * CC + h * HD;
      AF b0, b1;
      b0.u[0] = *(const uint4*)(kr + 16 * hi);      b0.u[1] = *(const uint4*)(kr + 16 * hi + 8);
      b1.u[0] = *(const uint4*)(kr + 32 + 16 * hi); b1.u[1] = *(const uint4*)(kr + 32 + 16 * hi + 8);
      v8f acc = WMMA_BF16(aq0.v, b0.v, z.v);
      s0.v = WMMA_BF16(aq1.v, b1.v, acc);
    }
    {
      const unsigned short* kr = Kb + ((size_t)(b * NN + k0 + 16 + ln)) * CC + h * HD;
      AF b0, b1;
      b0.u[0] = *(const uint4*)(kr + 16 * hi);      b0.u[1] = *(const uint4*)(kr + 16 * hi + 8);
      b1.u[0] = *(const uint4*)(kr + 32 + 16 * hi); b1.u[1] = *(const uint4*)(kr + 32 + 16 * hi + 8);
      v8f acc = WMMA_BF16(aq0.v, b0.v, z.v);
      s1.v = WMMA_BF16(aq1.v, b1.v, acc);
    }
    // scale 1/sqrt(64), online softmax per row (row = r + 8*hi, cols across 16-lane group)
#pragma unroll
    for (int r = 0; r < 8; r++) { s0.f[r] *= 0.125f; s1.f[r] *= 0.125f; }
#pragma unroll
    for (int r = 0; r < 8; r++) {
      float v = fmaxf(s0.f[r], s1.f[r]);
      for (int m = 1; m < 16; m <<= 1) v = fmaxf(v, __shfl_xor(v, m, 32));
      float nm = fmaxf(rmax[r], v);
      float p0 = __expf(s0.f[r] - nm);
      float p1 = __expf(s1.f[r] - nm);
      float corr = __expf(rmax[r] - nm);
      float ps = p0 + p1;
      for (int m = 1; m < 16; m <<= 1) ps += __shfl_xor(ps, m, 32);
      rsum[r] = rsum[r] * corr + ps;
      rmax[r] = nm;
#pragma unroll
      for (int d = 0; d < 4; d++) o[d].f[r] *= corr;
      int mrow = r + 8 * hi;                        // repack P (C-layout -> LDS)
      pbuf[w][mrow][ln]      = f2bf(p0);
      pbuf[w][mrow][16 + ln] = f2bf(p1);
    }
    asm volatile("s_wait_dscnt 0" ::: "memory");
    AF ap;                                           // A-layout read of P (16x32)
    const unsigned short* pr = &pbuf[w][ln][0];
    ap.u[0] = *(const uint4*)(pr + 16 * hi);
    ap.u[1] = *(const uint4*)(pr + 16 * hi + 8);
#pragma unroll
    for (int d4 = 0; d4 < 4; d4++) {
      const unsigned short* vr =
          Vt + (((size_t)(b * NHD + h)) * HD + d4 * 16 + ln) * NN + k0 + 16 * hi;
      AF bv;
      bv.u[0] = *(const uint4*)vr;
      bv.u[1] = *(const uint4*)(vr + 8);
      o[d4].v = WMMA_BF16(ap.v, bv.v, o[d4].v);
    }
  }
#pragma unroll
  for (int d4 = 0; d4 < 4; d4++) {
    int dcol = d4 * 16 + ln;
#pragma unroll
    for (int r = 0; r < 8; r++) {
      float val = o[d4].f[r] / rsum[r];
      int q = qt * 16 + r + 8 * hi;
      ctxb[((size_t)(b * NN + q)) * CC + h * HD + dcol] = f2bf(val);
    }
  }
}

// ---------------- dynamic 5x5 depthwise conv (kernel shared across channels) ----------------
__global__ void k_dyn(const float* fused, const float* kw, float* dyn) {
  int b = blockIdx.y;
  __shared__ float kk[25];
  if (threadIdx.x < 25) kk[threadIdx.x] = kw[b * 25 + threadIdx.x];
  __syncthreads();
  int idx = blockIdx.x * 256 + threadIdx.x;       // c*1024 + n
  int c = idx >> 10, n = idx & (NN - 1);
  int y = n >> 5, x = n & 31;
  const float* pf = fused + ((size_t)(b * CC + c)) * NN;
  float s = 0.f;
  for (int ky = 0; ky < 5; ky++) {
    int yy = y + ky - 2; if (yy < 0 || yy >= 32) continue;
    for (int kx = 0; kx < 5; kx++) {
      int xx = x + kx - 2; if (xx < 0 || xx >= 32) continue;
      s += kk[ky * 5 + kx] * pf[yy * 32 + xx];
    }
  }
  dyn[((size_t)(b * CC + c)) * NN + n] = s;
}

// ---------------- gate convs (scalar; small fraction of FLOPs) ----------------
__global__ void k_gate1(const float* vi, const float* ir, const float* w1,
                        const float* b1, float* out) {
  int b = blockIdx.z, oc = blockIdx.y;
  __shared__ float wsh[9216];                       // 1024 ic * 9
  const float* wp = w1 + (size_t)oc * 9216;
  for (int i = threadIdx.x; i < 9216; i += 256) wsh[i] = wp[i];
  __syncthreads();
  int n = blockIdx.x * 256 + threadIdx.x;
  int y = n >> 5, x = n & 31;
  float s = b1[oc];
  for (int ic = 0; ic < 1024; ic++) {
    const float* src = (ic < CC) ? vi + ((size_t)(b * CC + ic)) * NN
                                 : ir + ((size_t)(b * CC + ic - CC)) * NN;
    const float* wr = wsh + ic * 9;
    for (int ky = 0; ky < 3; ky++) {
      int yy = y + ky - 1; if (yy < 0 || yy >= 32) continue;
      for (int kx = 0; kx < 3; kx++) {
        int xx = x + kx - 1; if (xx < 0 || xx >= 32) continue;
        s += wr[ky * 3 + kx] * src[yy * 32 + xx];
      }
    }
  }
  out[((size_t)(b * 32 + oc)) * NN + n] = fmaxf(s, 0.f);
}

__global__ void k_gate2(const float* gh, const float* w2, const float* b2, float* gates) {
  int b = blockIdx.y;
  int idx = blockIdx.x * 256 + threadIdx.x;         // oc*1024+n, 2048 total
  if (idx >= 2048) return;
  int oc = idx >> 10, n = idx & (NN - 1);
  int y = n >> 5, x = n & 31;
  float s = b2[oc];
  for (int ic = 0; ic < 32; ic++) {
    const float* src = gh + ((size_t)(b * 32 + ic)) * NN;
    const float* wr = w2 + ((size_t)(oc * 32 + ic)) * 9;
    for (int ky = 0; ky < 3; ky++) {
      int yy = y + ky - 1; if (yy < 0 || yy >= 32) continue;
      for (int kx = 0; kx < 3; kx++) {
        int xx = x + kx - 1; if (xx < 0 || xx >= 32) continue;
        s += wr[ky * 3 + kx] * src[yy * 32 + xx];
      }
    }
  }
  gates[((size_t)b * 2) * NN + idx] = 1.f / (1.f + __expf(-s));
}

// fused = (w_vi + w_ir)*dyn + vi + ir  -> bf16 in (b,n,c) layout (channels contiguous,
// so the enh implicit-GEMM gets vectorized b128 A-fragment loads)
__global__ void k_fuse(const float* vi, const float* ir, const float* dyn,
                       const float* gates, unsigned short* fusedb) {
  size_t idx = (size_t)blockIdx.x * blockDim.x + threadIdx.x;
  if (idx >= (size_t)BB * CC * NN) return;
  int n = (int)(idx & (NN - 1));
  size_t t = idx >> 10; int c = (int)(t & (CC - 1)); int b = (int)(t >> 9);
  float g = gates[((size_t)(b * 2)) * NN + n] + gates[((size_t)(b * 2 + 1)) * NN + n];
  fusedb[((size_t)(b * NN + n)) * CC + c] = f2bf(g * dyn[idx] + vi[idx] + ir[idx]);
}

// ---------------- enh conv as implicit GEMM (M=8192 px, N=512, K=9*512) + BN + SiLU --------
// fusedb is (b,n,c): a 3x3 tap is a row shift, so A-fragments are contiguous b128 loads.
// Per wave: 32(M) x 64(N).
__global__ __launch_bounds__(256) void k_enh(const unsigned short* __restrict__ fusedb,
                                             const unsigned short* __restrict__ wpk,
                                             const float* __restrict__ enh_b,
                                             const float* __restrict__ gamma,
                                             const float* __restrict__ beta,
                                             const float* __restrict__ mean,
                                             const float* __restrict__ var,
                                             float* __restrict__ out) {
  int gw = (blockIdx.x * blockDim.x + threadIdx.x) >> 5;
  int lane = threadIdx.x & 31;
  int hi = lane >> 4, ln = lane & 15;
  int mt = gw >> 3, ng = gw & 7;                 // mt: 0..255 (32 pixels each)
  CF c[2][4];
#pragma unroll
  for (int h2 = 0; h2 < 2; h2++)
#pragma unroll
    for (int s = 0; s < 4; s++)
#pragma unroll
      for (int i = 0; i < 8; i++) c[h2][s].f[i] = 0.f;

  int pix0 = mt * 32 + ln;
  int pix1 = pix0 + 16;
  int b0p = pix0 >> 10, n0p = pix0 & (NN - 1);
  int b1p = pix1 >> 10, n1p = pix1 & (NN - 1);
  int y0 = n0p >> 5, x0 = n0p & 31;
  int y1 = n1p >> 5, x1 = n1p & 31;

  for (int tap = 0; tap < 9; tap++) {
    int dy = tap / 3 - 1, dx = tap % 3 - 1;
    int yy0 = y0 + dy, xx0 = x0 + dx;
    int yy1 = y1 + dy, xx1 = x1 + dx;
    bool v0 = (yy0 >= 0 && yy0 < 32 && xx0 >= 0 && xx0 < 32);
    bool v1 = (yy1 >= 0 && yy1 < 32 && xx1 >= 0 && xx1 < 32);
    const unsigned short* ar0 =
        fusedb + ((size_t)(b0p * NN + (v0 ? yy0 * 32 + xx0 : 0))) * CC;
    const unsigned short* ar1 =
        fusedb + ((size_t)(b1p * NN + (v1 ? yy1 * 32 + xx1 : 0))) * CC;
    const unsigned short* wtap = wpk + (size_t)tap * CC * CC;
    for (int icb = 0; icb < CC; icb += 32) {
      AF a0, a1;
      if (v0) {
        a0.u[0] = *(const uint4*)(ar0 + icb + 8 * hi);
        a0.u[1] = *(const uint4*)(ar0 + icb + 16 + 8 * hi);
      } else {
#pragma unroll
        for (int j = 0; j < 16; j++) a0.s[j] = 0;
      }
      if (v1) {
        a1.u[0] = *(const uint4*)(ar1 + icb + 8 * hi);
        a1.u[1] = *(const uint4*)(ar1 + icb + 16 + 8 * hi);
      } else {
#pragma unroll
        for (int j = 0; j < 16; j++) a1.s[j] = 0;
      }
#pragma unroll
      for (int s4 = 0; s4 < 4; s4++) {
        int col = ng * 64 + s4 * 16 + ln;
        const unsigned short* wr = wtap + (size_t)col * CC + icb + 8 * hi;
        AF bf;
        bf.u[0] = *(const uint4*)wr;
        bf.u[1] = *(const uint4*)(wr + 16);
        c[0][s4].v = WMMA_BF16(a0.v, bf.v, c[0][s4].v);
        c[1][s4].v = WMMA_BF16(a1.v, bf.v, c[1][s4].v);
      }
    }
  }

#pragma unroll
  for (int h2 = 0; h2 < 2; h2++) {
    int row0 = mt * 32 + h2 * 16 + 8 * hi;
    int ob = row0 >> 10, n0 = row0 & (NN - 1);
#pragma unroll
    for (int s4 = 0; s4 < 4; s4++) {
      int col = ng * 64 + s4 * 16 + ln;
      float sc = gamma[col] * rsqrtf(var[col] + 1e-5f);
      float bi = enh_b[col], mu = mean[col], bt = beta[col];
      CF t;
#pragma unroll
      for (int r = 0; r < 8; r++) {
        float yv = (c[h2][s4].f[r] + bi - mu) * sc + bt;
        t.f[r] = yv / (1.f + __expf(-yv));            // SiLU
      }
      float4* dst = (float4*)(out + ((size_t)(ob * CC + col)) * NN + n0);
      dst[0] = t.q[0]; dst[1] = t.q[1];
    }
  }
}

// ---------------- host orchestration ----------------
extern "C" void kernel_launch(void* const* d_in, const int* in_sizes, int n_in,
                              void* d_out, int out_size, void* d_ws, size_t ws_size,
                              hipStream_t stream) {
  const float* vi    = (const float*)d_in[0];
  const float* ir    = (const float*)d_in[1];
  const float* dc_w1 = (const float*)d_in[2];
  const float* dc_b1 = (const float*)d_in[3];
  const float* dc_w2 = (const float*)d_in[4];
  const float* dc_b2 = (const float*)d_in[5];
  const float* Wq    = (const float*)d_in[6];
  const float* bq    = (const float*)d_in[7];
  const float* Wk    = (const float*)d_in[8];
  const float* bk    = (const float*)d_in[9];
  const float* Wv    = (const float*)d_in[10];
  const float* bv    = (const float*)d_in[11];
  const float* Wo    = (const float*)d_in[12];
  const float* bo    = (const float*)d_in[13];
  const float* gw1   = (const float*)d_in[14];
  const float* gb1   = (const float*)d_in[15];
  const float* gw2   = (const float*)d_in[16];
  const float* gb2   = (const float*)d_in[17];
  const float* enw   = (const float*)d_in[18];
  const float* enb   = (const float*)d_in[19];
  const float* gma   = (const float*)d_in[20];
  const float* bta   = (const float*)d_in[21];
  const float* bmu   = (const float*)d_in[22];
  const float* bvar  = (const float*)d_in[23];
  float* out = (float*)d_out;

  // workspace carve-up (~98 MB total)
  char* ws = (char*)d_ws;
  size_t off = 0;
  auto alloc = [&](size_t bytes) -> char* {
    char* p = ws + off; off += (bytes + 255) & ~(size_t)255; return p;
  };
  unsigned short* Xvi  = (unsigned short*)alloc((size_t)MTOT * CC * 2);
  unsigned short* Xir  = (unsigned short*)alloc((size_t)MTOT * CC * 2);
  unsigned short* Wqb  = (unsigned short*)alloc((size_t)CC * CC * 2);
  unsigned short* Wkb  = (unsigned short*)alloc((size_t)CC * CC * 2);
  unsigned short* Wvb  = (unsigned short*)alloc((size_t)CC * CC * 2);
  unsigned short* Wob  = (unsigned short*)alloc((size_t)CC * CC * 2);
  unsigned short* Qb   = (unsigned short*)alloc((size_t)MTOT * CC * 2);
  unsigned short* Kb   = (unsigned short*)alloc((size_t)MTOT * CC * 2);
  unsigned short* Vt   = (unsigned short*)alloc((size_t)MTOT * CC * 2);
  unsigned short* ctxb = (unsigned short*)alloc((size_t)MTOT * CC * 2);
  float* fusedf        = (float*)alloc((size_t)BB * CC * NN * 4);
  float* dyn           = (float*)alloc((size_t)BB * CC * NN * 4);
  float* pooled        = (float*)alloc((size_t)BB * CC * 4);
  float* hid           = (float*)alloc((size_t)BB * 2048 * 4);
  float* kw            = (float*)alloc((size_t)BB * 25 * 4);
  float* gateh         = (float*)alloc((size_t)BB * 32 * NN * 4);
  float* gates         = (float*)alloc((size_t)BB * 2 * NN * 4);
  unsigned short* fsb  = (unsigned short*)alloc((size_t)BB * CC * NN * 2);
  unsigned short* wpk  = (unsigned short*)alloc((size_t)9 * CC * CC * 2);
  (void)ws_size; (void)in_sizes; (void)n_in; (void)out_size;

  // stage bf16 operands
  k_pack_seq<<<16384, 256, 0, stream>>>(vi, ir, Xvi, Xir);
  k_cvt<<<1024, 256, 0, stream>>>(Wq, Wqb, CC * CC);
  k_cvt<<<1024, 256, 0, stream>>>(Wk, Wkb, CC * CC);
  k_cvt<<<1024, 256, 0, stream>>>(Wv, Wvb, CC * CC);
  k_cvt<<<1024, 256, 0, stream>>>(Wo, Wob, CC * CC);
  k_pack_enhw<<<9216, 256, 0, stream>>>(enw, wpk);

  // dynamic kernel MLP
  k_pool<<<BB * CC, 256, 0, stream>>>(vi, ir, pooled);
  k_mlp1<<<64, 256, 0, stream>>>(pooled, dc_w1, dc_b1, hid);
  k_mlp2<<<BB, 32, 0, stream>>>(hid, dc_w2, dc_b2, kw);

  // QKV projections (WMMA, 32x64 tiles/wave)
  k_gemm<0><<<256, 256, 0, stream>>>(Xvi, Wqb, bq, (void*)Qb, MTOT);
  k_gemm<0><<<256, 256, 0, stream>>>(Xir, Wkb, bk, (void*)Kb, MTOT);
  k_gemm<1><<<256, 256, 0, stream>>>(Xir, Wvb, bv, (void*)Vt, MTOT);

  // flash attention (WMMA)
  k_attn<<<1024, 128, 0, stream>>>(Qb, Kb, Vt, ctxb);

  // output projection -> fp32 spatial fused_feat (WMMA)
  k_gemm<2><<<256, 256, 0, stream>>>(ctxb, Wob, bo, (void*)fusedf, MTOT);

  // dynamic depthwise conv + gates + fuse
  k_dyn<<<dim3(2048, BB), 256, 0, stream>>>(fusedf, kw, dyn);
  k_gate1<<<dim3(4, 32, BB), 256, 0, stream>>>(vi, ir, gw1, gb1, gateh);
  k_gate2<<<dim3(8, BB), 256, 0, stream>>>(gateh, gw2, gb2, gates);
  k_fuse<<<16384, 256, 0, stream>>>(vi, ir, dyn, gates, fsb);

  // enh 3x3 conv (implicit GEMM, WMMA, 32x64 tiles/wave) + BN + SiLU -> d_out
  k_enh<<<256, 256, 0, stream>>>(fsb, wpk, enb, gma, bta, bmu, bvar, out);
}
